// NDCN1_56324201120323
// MI455X (gfx1250) — matrix operand order repaired and checked by reading, not compiled, not run on hardware
//
#include <hip/hip_runtime.h>
#include <hip/hip_bf16.h>
#include <math.h>

// ---------------------------------------------------------------------------
// NDCN forward on MI455X (gfx1250).
// Dominant cost: 19x  z = A[10000x10000] @ y[10000x32]  (memory bound on A).
// Strategy: convert A to bf16 once (halves HBM traffic per step: 200MB/step),
// keep y transposed in bf16 so both WMMA operands load as contiguous b128,
// accumulate in fp32 via v_wmma_f32_16x16x32_bf16, fuse the small [32x32]
// GEMMs (Wt / Wo), relu, Euler update and tanh output epilogue in fp32.
// A is streamed with non-temporal loads so the L2 keeps yT + weights hot.
// ---------------------------------------------------------------------------

#define NN    10000          // nodes
#define HID   32             // hidden == classes == input size
#define TT    20             // time ticks
#define KPAD  10016          // NN padded up to a multiple of 32 (313 chunks)
#define KCH   (KPAD / 32)    // 313 K-chunks of 32

typedef __bf16 bf16_t;
typedef __attribute__((ext_vector_type(16))) __bf16 v16bf;
typedef __attribute__((ext_vector_type(8)))  __bf16 v8bf;
typedef __attribute__((ext_vector_type(8)))  float  v8f;
typedef __attribute__((ext_vector_type(4)))  float  v4f;

__device__ __forceinline__ bf16_t f2bf(float f) {
    unsigned u = __builtin_bit_cast(unsigned, f);
    unsigned r = u + 0x7FFFu + ((u >> 16) & 1u);        // round-to-nearest-even
    unsigned short h = (unsigned short)(r >> 16);
    return __builtin_bit_cast(bf16_t, h);
}

// ---------------------------------------------------------------------------
// A (fp32, [NN x NN]) -> Ab (bf16, [NN x KPAD], zero padded columns)
// grid: (ceil((KPAD/8)/256), NN), block 256; each thread converts 8 elements.
// Both streams are one-shot: non-temporal load + store.
// ---------------------------------------------------------------------------
__global__ void ndcn_cvtA(const float* __restrict__ A, bf16_t* __restrict__ Ab) {
    const int row = blockIdx.y;
    const int cg  = blockIdx.x * blockDim.x + threadIdx.x;   // 8-col group
    if (cg >= KPAD / 8) return;
    const int col = cg * 8;
    v8bf o;
    if (col < NN) {
        const v4f* p = (const v4f*)(A + (size_t)row * NN + col);
        v4f f0 = __builtin_nontemporal_load(p);
        v4f f1 = __builtin_nontemporal_load(p + 1);
        o[0] = f2bf(f0.x); o[1] = f2bf(f0.y); o[2] = f2bf(f0.z); o[3] = f2bf(f0.w);
        o[4] = f2bf(f1.x); o[5] = f2bf(f1.y); o[6] = f2bf(f1.z); o[7] = f2bf(f1.w);
    } else {
        bf16_t z = f2bf(0.0f);
#pragma unroll
        for (int i = 0; i < 8; ++i) o[i] = z;
    }
    __builtin_nontemporal_store(o, (v8bf*)(Ab + (size_t)row * KPAD + col));
}

// ---------------------------------------------------------------------------
// Initial state: x = tanh(x0@Wp^T+bp); h = tanh(x@W1^T+b1)@W2^T+b2
// Writes y0 (fp32 row-major), yT0 (bf16, transposed [32 x KPAD]),
// out[0] = tanh(h@Wo^T+bo).  Also zeroes the K-pad columns of both yT buffers.
// One thread per node (grid covers KPAD so pad region gets zeroed).
// ---------------------------------------------------------------------------
__global__ void ndcn_init(const float* __restrict__ x0,
                          const float* __restrict__ Wp, const float* __restrict__ bp,
                          const float* __restrict__ W1, const float* __restrict__ b1,
                          const float* __restrict__ W2, const float* __restrict__ b2,
                          const float* __restrict__ Wo, const float* __restrict__ bo,
                          float* __restrict__ y0, bf16_t* __restrict__ yT0,
                          bf16_t* __restrict__ yT1, float* __restrict__ out0) {
    const int r = blockIdx.x * blockDim.x + threadIdx.x;
    if (r >= KPAD) return;
    if (r >= NN) {   // zero K-pad columns of both transposed buffers
        bf16_t z = f2bf(0.0f);
#pragma unroll
        for (int j = 0; j < HID; ++j) {
            yT0[(size_t)j * KPAD + r] = z;
            yT1[(size_t)j * KPAD + r] = z;
        }
        return;
    }
    float a[HID], b[HID];
#pragma unroll
    for (int k = 0; k < HID; ++k) a[k] = x0[(size_t)r * HID + k];
#pragma unroll 4
    for (int j = 0; j < HID; ++j) {            // preprocess: tanh(Wp a + bp)
        float s = bp[j];
#pragma unroll
        for (int k = 0; k < HID; ++k) s = fmaf(Wp[j * HID + k], a[k], s);
        b[j] = tanhf(s);
    }
#pragma unroll 4
    for (int j = 0; j < HID; ++j) {            // tanh(W1 b + b1)
        float s = b1[j];
#pragma unroll
        for (int k = 0; k < HID; ++k) s = fmaf(W1[j * HID + k], b[k], s);
        a[j] = tanhf(s);
    }
#pragma unroll 4
    for (int j = 0; j < HID; ++j) {            // h = W2 a + b2
        float s = b2[j];
#pragma unroll
        for (int k = 0; k < HID; ++k) s = fmaf(W2[j * HID + k], a[k], s);
        b[j] = s;
    }
#pragma unroll
    for (int j = 0; j < HID; ++j) {
        y0[(size_t)r * HID + j]   = b[j];
        yT0[(size_t)j * KPAD + r] = f2bf(b[j]);
    }
#pragma unroll 4
    for (int j = 0; j < HID; ++j) {            // out[0] = tanh(Wo h + bo)
        float s = bo[j];
#pragma unroll
        for (int k = 0; k < HID; ++k) s = fmaf(Wo[j * HID + k], b[k], s);
        out0[(size_t)r * HID + j] = tanhf(s);
    }
}

// ---------------------------------------------------------------------------
// One Euler step. One wave per 16-row strip of the output (625 blocks x 32).
//   z = A @ y                      (WMMA bf16 -> fp32, K-loop over 313 chunks)
//   w = relu(z @ Wt^T + bt)        (fp32 scalar via LDS)
//   y' = y + dt * w                (dt = vt[t+1]-vt[t], read on device)
//   out[t+1] = tanh(y' @ Wo^T+bo)  (fused output layer)
// kch is a RUNTIME arg so the K-loop cannot be fully unrolled (keeps the hot
// loop resident in I$); unroll-4 batches 24 b128 loads ahead of 8 WMMAs.
// ---------------------------------------------------------------------------
__global__ __launch_bounds__(32)
void ndcn_step(const bf16_t* __restrict__ Ab, const bf16_t* __restrict__ yTin,
               const float* __restrict__ yin,
               const float* __restrict__ Wt, const float* __restrict__ bt,
               const float* __restrict__ Wo, const float* __restrict__ bo,
               const float* __restrict__ vt, int tstep, int kch,
               float* __restrict__ yout, bf16_t* __restrict__ yTout,
               float* __restrict__ outslice) {
    const int lane = threadIdx.x;          // 0..31
    const int m0   = blockIdx.x * 16;      // output row base (625*16 == 10000)
    const int mrow = lane & 15;
    const bool hi  = lane >= 16;

    // A fragment (16x32 bf16): lanes 0-15 carry K {0..7,16..23}+32c of row
    // m0+mrow; lanes 16-31 carry K {8..15,24..31}+32c.  (ISA 7.12.2 tables)
    const bf16_t* __restrict__ arow = Ab + (size_t)(m0 + mrow) * KPAD;
    const int aoff = hi ? 8 : 0;
    // B fragments (32x16 bf16): lane group 0-15 holds K {0..7,16..23} of column
    // n = mrow (+16 for second tile); lanes 16-31 hold K {8..15,24..31}.
    const bf16_t* __restrict__ b0row = yTin + (size_t)mrow * KPAD;
    const bf16_t* __restrict__ b1row = yTin + (size_t)(mrow + 16) * KPAD;
    const int boff = hi ? 8 : 0;

    v8f acc0 = {};   // C tile, output cols 0..15
    v8f acc1 = {};   // C tile, output cols 16..31

#pragma unroll 4
    for (int c = 0; c < kch; ++c) {
        const int kb = c * 32;
        // A: streaming, zero reuse within a step and >L2 across steps -> NT.
        v8bf alo  = __builtin_nontemporal_load((const v8bf*)(arow + kb + aoff));
        v8bf ahi  = __builtin_nontemporal_load((const v8bf*)(arow + kb + 16 + aoff));
        // y^T: 640KB, re-read by all 625 waves -> keep regular temporal (L2-hot).
        v8bf b0lo = *(const v8bf*)(b0row + kb + boff);
        v8bf b0hi = *(const v8bf*)(b0row + kb + boff + 16);
        v8bf b1lo = *(const v8bf*)(b1row + kb + boff);
        v8bf b1hi = *(const v8bf*)(b1row + kb + boff + 16);

        v16bf a  = __builtin_shufflevector(alo,  ahi,
                     0,1,2,3,4,5,6,7,8,9,10,11,12,13,14,15);
        v16bf b0 = __builtin_shufflevector(b0lo, b0hi,
                     0,1,2,3,4,5,6,7,8,9,10,11,12,13,14,15);
        v16bf b1 = __builtin_shufflevector(b1lo, b1hi,
                     0,1,2,3,4,5,6,7,8,9,10,11,12,13,14,15);

        acc0 = __builtin_amdgcn_wmma_f32_16x16x32_bf16(
                   false, a, false, b0, (short)0, acc0, false, false);
        acc1 = __builtin_amdgcn_wmma_f32_16x16x32_bf16(
                   false, a, false, b1, (short)0, acc1, false, false);
    }

    // --- epilogue: spill z to LDS (C layout: lanes 0-15 rows 0-7, col=lane;
    //     lanes 16-31 rows 8-15, col=lane-16) ---
    __shared__ float zs[16][33];
    __shared__ float ys[16][33];
    const int rbase = hi ? 8 : 0;
#pragma unroll
    for (int r = 0; r < 8; ++r) {
        zs[rbase + r][mrow]      = acc0[r];
        zs[rbase + r][mrow + 16] = acc1[r];
    }
    __syncthreads();   // single-wave workgroup: S_NOP + compiler LDS fence

    const float dt = vt[tstep + 1] - vt[tstep];
    const int j = lane;                       // this lane owns output column j
    float wtj[HID], woj[HID];
#pragma unroll
    for (int k = 0; k < HID; ++k) {
        wtj[k] = Wt[j * HID + k];
        woj[k] = Wo[j * HID + k];
    }
    const float btj = bt[j], boj = bo[j];

    float yn[16];
#pragma unroll
    for (int m = 0; m < 16; ++m) {
        float s = btj;
#pragma unroll
        for (int k = 0; k < HID; ++k) s = fmaf(zs[m][k], wtj[k], s);
        float w = s > 0.0f ? s : 0.0f;                       // relu
        float v = yin[(size_t)(m0 + m) * HID + j] + dt * w;  // Euler
        yn[m] = v;
        yout[(size_t)(m0 + m) * HID + j] = v;
        ys[m][j] = v;
    }
    __syncthreads();

    // transposed bf16 y' for next step: lane j owns yT row j, 16 contiguous.
    v8bf p0, p1;
#pragma unroll
    for (int m = 0; m < 8; ++m) { p0[m] = f2bf(yn[m]); p1[m] = f2bf(yn[m + 8]); }
    *(v8bf*)(yTout + (size_t)j * KPAD + m0)     = p0;
    *(v8bf*)(yTout + (size_t)j * KPAD + m0 + 8) = p1;

    // fused output layer: out[t+1] = tanh(y' @ Wo^T + bo); never re-read -> NT.
#pragma unroll
    for (int m = 0; m < 16; ++m) {
        float s = boj;
#pragma unroll
        for (int k = 0; k < HID; ++k) s = fmaf(ys[m][k], woj[k], s);
        __builtin_nontemporal_store(tanhf(s),
            outslice + (size_t)(m0 + m) * HID + j);
    }
}

// ---------------------------------------------------------------------------
extern "C" void kernel_launch(void* const* d_in, const int* in_sizes, int n_in,
                              void* d_out, int out_size, void* d_ws, size_t ws_size,
                              hipStream_t stream) {
    const float* vt = (const float*)d_in[0];
    const float* x0 = (const float*)d_in[1];
    const float* A  = (const float*)d_in[2];
    const float* Wp = (const float*)d_in[3];
    const float* bp = (const float*)d_in[4];
    const float* W1 = (const float*)d_in[5];
    const float* b1 = (const float*)d_in[6];
    const float* W2 = (const float*)d_in[7];
    const float* b2 = (const float*)d_in[8];
    const float* Wt = (const float*)d_in[9];
    const float* bt = (const float*)d_in[10];
    const float* Wo = (const float*)d_in[11];
    const float* bo = (const float*)d_in[12];
    float* out = (float*)d_out;

    // workspace layout (all chunks 256B-aligned by construction):
    //   Ab   : NN*KPAD bf16  = 200,320,000 B
    //   yT0/1: 32*KPAD bf16  =     641,024 B each
    //   y0/1 : NN*32  fp32   =   1,280,000 B each
    char* w = (char*)d_ws;
    bf16_t* Ab  = (bf16_t*)w;  w += (size_t)NN * KPAD * sizeof(bf16_t);
    bf16_t* yT0 = (bf16_t*)w;  w += (size_t)HID * KPAD * sizeof(bf16_t);
    bf16_t* yT1 = (bf16_t*)w;  w += (size_t)HID * KPAD * sizeof(bf16_t);
    float*  y0  = (float*)w;   w += (size_t)NN * HID * sizeof(float);
    float*  y1  = (float*)w;

    // 1) A -> bf16 (padded)
    dim3 cgrid((KPAD / 8 + 255) / 256, NN, 1);
    ndcn_cvtA<<<cgrid, 256, 0, stream>>>(A, Ab);

    // 2) initial state + out[0] + pad zeroing
    ndcn_init<<<(KPAD + 255) / 256, 256, 0, stream>>>(
        x0, Wp, bp, W1, b1, W2, b2, Wo, bo, y0, yT0, yT1, out);

    // 3) 19 Euler steps (ping-pong y buffers), each fuses the output layer
    bf16_t* yTbuf[2] = { yT0, yT1 };
    float*  ybuf[2]  = { y0,  y1  };
    for (int t = 0; t < TT - 1; ++t) {
        ndcn_step<<<NN / 16, 32, 0, stream>>>(
            Ab, yTbuf[t & 1], ybuf[t & 1], Wt, bt, Wo, bo, vt, t, KCH,
            ybuf[(t + 1) & 1], yTbuf[(t + 1) & 1],
            out + (size_t)(t + 1) * NN * HID);
    }
}
